// VQQuantizer_20031727468686
// MI455X (gfx1250) — compile-verified, matrix operand order checked by reading, not run on hardware
//
#include <hip/hip_runtime.h>
#include <hip/hip_bf16.h>

#define B_SIZE 16384
#define K_SIZE 4096
#define D_SIZE 512
#define TAU    1.0f
#define BETA   0.25f

typedef _Float16 v4h  __attribute__((ext_vector_type(4)));
typedef _Float16 v8h  __attribute__((ext_vector_type(8)));
typedef _Float16 v16h __attribute__((ext_vector_type(16)));
typedef float    v8f  __attribute__((ext_vector_type(8)));

static __device__ __forceinline__ v8f wmma_f16(v16h a, v16h b, v8f c) {
  // D(16x16,f32) = A(16x32,f16) x B(32x16,f16) + C
  return __builtin_amdgcn_wmma_f32_16x16x32_f16(false, a, false, b, (short)0, c,
                                                false, false);
}

static __device__ __forceinline__ v8f vzero8() {
  v8f z;
#pragma unroll
  for (int i = 0; i < 8; ++i) z[i] = 0.0f;
  return z;
}

// Load a 16x32 f16 tile in the CDNA5 WMMA 16-bit A-operand layout from
// row-major global memory (B-operand uses the same pattern applied to B^T,
// i.e. row-major rows of the "other" matrix).
// Per ISA 7.12.2: lane m (0..15) holds K = {hh..hh+7, hh+16..hh+23}, hh = (lane>>4)*8.
static __device__ __forceinline__ v16h load_tile_g(const _Float16* p, size_t ld,
                                                   int row0, int k0) {
  int lane = threadIdx.x & 31;
  int m  = lane & 15;
  int hh = (lane >> 4) * 8;
  const _Float16* base = p + (size_t)(row0 + m) * ld + (size_t)(k0 + hh);
  v8h lo = *(const v8h*)base;          // K = hh .. hh+7   (global_load_b128)
  v8h hi = *(const v8h*)(base + 16);   // K = hh+16 .. hh+23
  v16h r;
#pragma unroll
  for (int i = 0; i < 8; ++i) { r[i] = lo[i]; r[i + 8] = hi[i]; }
  return r;
}

// Same tile load but from LDS (row stride fixed at 32 halves) -> ds_load_b128.
static __device__ __forceinline__ v16h load_tile_lds(const _Float16* p, int row0) {
  int lane = threadIdx.x & 31;
  int m  = lane & 15;
  int hh = (lane >> 4) * 8;
  const _Float16* base = p + (row0 + m) * 32 + hh;
  v8h lo = *(const v8h*)base;
  v8h hi = *(const v8h*)(base + 16);
  v16h r;
#pragma unroll
  for (int i = 0; i < 8; ++i) { r[i] = lo[i]; r[i + 8] = hi[i]; }
  return r;
}

// ---------------- prep: f32 -> f16 staging ----------------

__global__ __launch_bounds__(256) void conv_h_kernel(const float* __restrict__ h,
                                                     _Float16* __restrict__ h16) {
  int i = (blockIdx.x * 256 + threadIdx.x) * 4;
  float4 v = *(const float4*)(h + i);
  v4h o; o[0] = (_Float16)v.x; o[1] = (_Float16)v.y;
  o[2] = (_Float16)v.z; o[3] = (_Float16)v.w;
  *(v4h*)(h16 + i) = o;
}

// One wave per codebook row: emit cb16 (row-major), cbT16 (transposed, so
// GEMM2 B-operand loads are contiguous), and c_sq[k] via wave reduction.
__global__ __launch_bounds__(256) void prep_cb_kernel(const float* __restrict__ cb,
                                                      _Float16* __restrict__ cb16,
                                                      _Float16* __restrict__ cbT16,
                                                      float* __restrict__ csq) {
  int wave = threadIdx.x >> 5, lane = threadIdx.x & 31;
  int k = blockIdx.x * 8 + wave;
  const float* row = cb + (size_t)k * D_SIZE;
  int d0 = lane * 16;
  float acc = 0.0f;
#pragma unroll
  for (int j = 0; j < 4; ++j) {
    float4 v = *(const float4*)(row + d0 + j * 4);
    v4h o; o[0] = (_Float16)v.x; o[1] = (_Float16)v.y;
    o[2] = (_Float16)v.z; o[3] = (_Float16)v.w;
    *(v4h*)(cb16 + (size_t)k * D_SIZE + d0 + j * 4) = o;
    int d = d0 + j * 4;
    cbT16[(size_t)(d + 0) * K_SIZE + k] = o[0];
    cbT16[(size_t)(d + 1) * K_SIZE + k] = o[1];
    cbT16[(size_t)(d + 2) * K_SIZE + k] = o[2];
    cbT16[(size_t)(d + 3) * K_SIZE + k] = o[3];
    acc += v.x * v.x + v.y * v.y + v.z * v.z + v.w * v.w;
  }
#pragma unroll
  for (int o = 16; o >= 1; o >>= 1) acc += __shfl_xor(acc, o);
  if (lane == 0) csq[k] = acc;
}

// ---------------- GEMM1: logits s = (2*h.cb^T - c_sq + g)/tau ----------------
// 8 waves per WG; WG tile = 128 rows x 128 codes; codebook slab staged in LDS.

__global__ __launch_bounds__(256) void gemm1_kernel(const _Float16* __restrict__ h16,
                                                    const _Float16* __restrict__ cb16,
                                                    const float* __restrict__ csq,
                                                    const float* __restrict__ gum,
                                                    float* __restrict__ qbuf) {
  __shared__ _Float16 bLDS[128 * 32];  // 8 KB: codes 0..127 of this strip, 32 k's
  int tid = threadIdx.x;
  int wave = tid >> 5, lane = tid & 31;
  int row0 = (blockIdx.x * 8 + wave) * 16;
  int n0 = blockIdx.y * 128;

  v8f acc[8];
#pragma unroll
  for (int t = 0; t < 8; ++t) acc[t] = vzero8();

  int ldsRow = tid >> 1;
  int ldsOff = (tid & 1) * 16;

  for (int kd = 0; kd < D_SIZE; kd += 32) {
    // cooperative stage of 128x32 codebook slab (shared by all 8 row-waves)
    const _Float16* src = cb16 + (size_t)(n0 + ldsRow) * D_SIZE + kd + ldsOff;
    v8h s0 = *(const v8h*)src;
    v8h s1 = *(const v8h*)(src + 8);
    *(v8h*)(bLDS + ldsRow * 32 + ldsOff)     = s0;
    *(v8h*)(bLDS + ldsRow * 32 + ldsOff + 8) = s1;
    __syncthreads();

    v16h A = load_tile_g(h16, D_SIZE, row0, kd);
    if (kd + 32 < D_SIZE)  // hint next A slab into cache
      __builtin_prefetch(h16 + (size_t)(row0 + (lane & 15)) * D_SIZE + kd + 32, 0, 0);
#pragma unroll
    for (int nt = 0; nt < 8; ++nt) {
      v16h Bt = load_tile_lds(bLDS, nt * 16);
      acc[nt] = wmma_f16(A, Bt, acc[nt]);
    }
    __syncthreads();
  }

  // Epilogue. C layout: lane l -> n = l&15, m = 8*(l>>4) + r.
  // h_sq is a per-row constant -> dropped (softmax invariant).
  int n = lane & 15, g = lane >> 4;
#pragma unroll
  for (int nt = 0; nt < 8; ++nt) {
    int nG = n0 + nt * 16 + n;
    float cs = csq[nG];
#pragma unroll
    for (int r = 0; r < 8; ++r) {
      int bG = row0 + g * 8 + r;
      size_t o = (size_t)bG * K_SIZE + nG;
      qbuf[o] = (2.0f * acc[nt][r] - cs + gum[o]) * (1.0f / TAU);
    }
  }
}

// ---------------- row stats: max, argmax (first index), sum(exp) ----------------

__global__ __launch_bounds__(256) void rowstat_kernel(const float* __restrict__ qbuf,
                                                      float* __restrict__ rowmax,
                                                      float* __restrict__ rinv,
                                                      int* __restrict__ amax) {
  __shared__ float sv[256];
  __shared__ int   si[256];
  __shared__ float ss[256];
  int b = blockIdx.x, tid = threadIdx.x;
  const float* row = qbuf + (size_t)b * K_SIZE;

  float mx = -3.4e38f; int ai = tid;
  for (int i = tid; i < K_SIZE; i += 256) {
    float v = row[i];
    if (v > mx) { mx = v; ai = i; }
  }
  sv[tid] = mx; si[tid] = ai;
  for (int off = 128; off > 0; off >>= 1) {
    __syncthreads();
    if (tid < off) {
      float v2 = sv[tid + off]; int i2 = si[tid + off];
      if (v2 > sv[tid] || (v2 == sv[tid] && i2 < si[tid])) { sv[tid] = v2; si[tid] = i2; }
    }
  }
  __syncthreads();
  float M = sv[0];

  float s = 0.0f;
  for (int i = tid; i < K_SIZE; i += 256) s += expf(row[i] - M);
  ss[tid] = s;
  for (int off = 128; off > 0; off >>= 1) {
    __syncthreads();
    if (tid < off) ss[tid] += ss[tid + off];
  }
  if (tid == 0) { rowmax[b] = M; rinv[b] = 1.0f / ss[0]; amax[b] = si[0]; }
}

// ---------------- GEMM2 fused with softmax: q (in place) and c_tilde ----------------
// One WG per 16 rows. Each k-step: read logits once, write final q, stage e(f16)
// in LDS as the WMMA A-operand; B from cbT16 (4 MB, L2-resident on 192 MB L2).

__global__ __launch_bounds__(256) void gemm2_kernel(const _Float16* __restrict__ cbT16,
                                                    const float* __restrict__ rowmax,
                                                    const float* __restrict__ rinv,
                                                    float* __restrict__ qbuf,
                                                    float* __restrict__ c_tilde) {
  __shared__ _Float16 eLDS[16 * 32];  // 1 KB e-tile
  __shared__ float rm[16], ri[16];
  int tid = threadIdx.x, wave = tid >> 5, lane = tid & 31;
  int row0 = blockIdx.x * 16;
  if (tid < 16) { rm[tid] = rowmax[row0 + tid]; ri[tid] = rinv[row0 + tid]; }
  __syncthreads();

  v8f acc[4];
#pragma unroll
  for (int t = 0; t < 4; ++t) acc[t] = vzero8();

  int d0 = wave * 64;  // each wave owns 64 of the 512 output dims

  for (int k0 = 0; k0 < K_SIZE; k0 += 32) {
    // produce: 512 elements, 2 per thread (coalesced along k)
    {
      int m = tid >> 5, kk = tid & 31;
      size_t o = (size_t)(row0 + m) * K_SIZE + k0 + kk;
      float e = expf(qbuf[o] - rm[m]);
      qbuf[o] = e * ri[m];                 // final normalized q, in place
      eLDS[m * 32 + kk] = (_Float16)e;
      int m2 = m + 8;
      size_t o2 = (size_t)(row0 + m2) * K_SIZE + k0 + kk;
      float e2 = expf(qbuf[o2] - rm[m2]);
      qbuf[o2] = e2 * ri[m2];
      eLDS[m2 * 32 + kk] = (_Float16)e2;
    }
    __syncthreads();

    v16h A = load_tile_lds(eLDS, 0);
#pragma unroll
    for (int nt = 0; nt < 4; ++nt) {
      v16h Bt = load_tile_g(cbT16, K_SIZE, d0 + nt * 16, k0);
      acc[nt] = wmma_f16(A, Bt, acc[nt]);
    }
    __syncthreads();  // protect eLDS before next overwrite
  }

  int n = lane & 15, g = lane >> 4;
#pragma unroll
  for (int nt = 0; nt < 4; ++nt) {
    int dG = d0 + nt * 16 + n;
#pragma unroll
    for (int r = 0; r < 8; ++r) {
      int m = g * 8 + r;
      c_tilde[(size_t)(row0 + m) * D_SIZE + dG] = acc[nt][r] * ri[m];
    }
  }
}

// ---------------- hard path: gather, c_quantized, per-row loss partials ----------------

__global__ __launch_bounds__(256) void hard_kernel(const float* __restrict__ h,
                                                   const float* __restrict__ cb,
                                                   const float* __restrict__ c_tilde,
                                                   const int* __restrict__ amax,
                                                   float* __restrict__ c_hard,
                                                   float* __restrict__ c_q,
                                                   float* __restrict__ rowloss) {
  int wave = threadIdx.x >> 5, lane = threadIdx.x & 31;
  int b = blockIdx.x * 8 + wave;
  int id = amax[b];
  const float* cr = cb + (size_t)id * D_SIZE;
  const float* hr = h + (size_t)b * D_SIZE;
  const float* tr = c_tilde + (size_t)b * D_SIZE;
  float* hd = c_hard + (size_t)b * D_SIZE;
  float* qd = c_q + (size_t)b * D_SIZE;
  int d0 = lane * 16;
  float acc = 0.0f;
#pragma unroll
  for (int j = 0; j < 4; ++j) {
    float4 cv = *(const float4*)(cr + d0 + j * 4);
    float4 hv = *(const float4*)(hr + d0 + j * 4);
    float4 tv = *(const float4*)(tr + d0 + j * 4);
    *(float4*)(hd + d0 + j * 4) = cv;
    float4 qv;  // straight-through forward: c_tilde + (c_hard - c_tilde)
    qv.x = tv.x + (cv.x - tv.x); qv.y = tv.y + (cv.y - tv.y);
    qv.z = tv.z + (cv.z - tv.z); qv.w = tv.w + (cv.w - tv.w);
    *(float4*)(qd + d0 + j * 4) = qv;
    float dx = hv.x - cv.x, dy = hv.y - cv.y, dz = hv.z - cv.z, dw = hv.w - cv.w;
    acc += dx * dx + dy * dy + dz * dz + dw * dw;
  }
#pragma unroll
  for (int o = 16; o >= 1; o >>= 1) acc += __shfl_xor(acc, o);
  if (lane == 0)
    rowloss[b] = acc * ((1.0f + BETA) / (float)((size_t)B_SIZE * D_SIZE));
}

__global__ __launch_bounds__(256) void loss_kernel(const float* __restrict__ rowloss,
                                                   float* __restrict__ loss) {
  __shared__ float sm[256];
  float a = 0.0f;
  for (int i = threadIdx.x; i < B_SIZE; i += 256) a += rowloss[i];
  sm[threadIdx.x] = a;
  for (int off = 128; off > 0; off >>= 1) {
    __syncthreads();
    if (threadIdx.x < off) sm[threadIdx.x] += sm[threadIdx.x + off];
  }
  if (threadIdx.x == 0) *loss = sm[0];
}

// ---------------- launch ----------------

extern "C" void kernel_launch(void* const* d_in, const int* in_sizes, int n_in,
                              void* d_out, int out_size, void* d_ws, size_t ws_size,
                              hipStream_t stream) {
  const float* h   = (const float*)d_in[0];
  const float* cb  = (const float*)d_in[1];
  const float* gum = (const float*)d_in[2];

  float* out = (float*)d_out;
  float* q       = out;
  float* c_tilde = q + (size_t)B_SIZE * K_SIZE;
  float* c_hard  = c_tilde + (size_t)B_SIZE * D_SIZE;
  float* c_q     = c_hard + (size_t)B_SIZE * D_SIZE;
  float* loss    = c_q + (size_t)B_SIZE * D_SIZE;

  char* ws = (char*)d_ws;
  _Float16* h16   = (_Float16*)ws; ws += (size_t)B_SIZE * D_SIZE * 2;  // 16 MB
  _Float16* cb16  = (_Float16*)ws; ws += (size_t)K_SIZE * D_SIZE * 2;  // 4 MB
  _Float16* cbT16 = (_Float16*)ws; ws += (size_t)K_SIZE * D_SIZE * 2;  // 4 MB
  float* csq      = (float*)ws;    ws += (size_t)K_SIZE * 4;
  float* rowmax   = (float*)ws;    ws += (size_t)B_SIZE * 4;
  float* rinv     = (float*)ws;    ws += (size_t)B_SIZE * 4;
  int*   amax     = (int*)ws;      ws += (size_t)B_SIZE * 4;
  float* rowloss  = (float*)ws;

  conv_h_kernel<<<(B_SIZE * D_SIZE) / (256 * 4), 256, 0, stream>>>(h, h16);
  prep_cb_kernel<<<K_SIZE / 8, 256, 0, stream>>>(cb, cb16, cbT16, csq);
  gemm1_kernel<<<dim3(B_SIZE / 128, K_SIZE / 128), 256, 0, stream>>>(h16, cb16, csq, gum, q);
  rowstat_kernel<<<B_SIZE, 256, 0, stream>>>(q, rowmax, rinv, amax);
  gemm2_kernel<<<B_SIZE / 16, 256, 0, stream>>>(cbT16, rowmax, rinv, q, c_tilde);
  hard_kernel<<<B_SIZE / 8, 256, 0, stream>>>(h, cb, c_tilde, amax, c_hard, c_q, rowloss);
  loss_kernel<<<1, 256, 0, stream>>>(rowloss, loss);
}
